// MAB_59382217835098
// MI455X (gfx1250) — matrix operand order repaired and checked by reading, not compiled, not run on hardware
//
#include <hip/hip_runtime.h>
#include <stdint.h>

typedef __attribute__((ext_vector_type(16))) __bf16 v16bf;
typedef __attribute__((ext_vector_type(8)))  __bf16 v8bf;
typedef __attribute__((ext_vector_type(4)))  __bf16 v4bf;
typedef __attribute__((ext_vector_type(8)))  float  v8f;

#define DIM 1024
#define NB  8
#define NSEQ 1024
#define NH  16
#define DH  64

static __device__ __forceinline__ v8f wmma_bf16(v16bf a, v16bf b, v8f c) {
  return __builtin_amdgcn_wmma_f32_16x16x32_bf16(false, a, false, b, (short)0, c, false, false);
}

// A-matrix 16x32 bf16 fragment (M x K), M striped over lanes, K packed in VGPRs.
// lanes 0-15: row=lane, K 0..7 at +0B, K 16..23 at +32B
// lanes16-31: row=lane-16, K 8..15 at +16B, K 24..31 at +48B
static __device__ __forceinline__ v16bf load_frag_a(const __bf16* p0, int stride) {
  int l = threadIdx.x & 15;
  int h = (threadIdx.x >> 4) & 1;
  const __bf16* p = p0 + l * stride + h * 8;
  v8bf lo = *(const v8bf*)p;
  v8bf hi = *(const v8bf*)(p + 16);
  return __builtin_shufflevector(lo, hi, 0,1,2,3,4,5,6,7,8,9,10,11,12,13,14,15);
}

// B-matrix 32x16 bf16 fragment (K x N), N striped over lanes.
// lanes 0-15: col=lane, K 0..7 and K 16..23; lanes 16-31: col=lane-16, K 8..15 and K 24..31
static __device__ __forceinline__ v16bf load_frag_b(const __bf16* p0, int stride) {
  int l = threadIdx.x & 15;
  int h = (threadIdx.x >> 4) & 1;
  const __bf16* p = p0 + l * stride + h * 8;
  v8bf lo = *(const v8bf*)p;
  v8bf hi = *(const v8bf*)(p + 16);
  return __builtin_shufflevector(lo, hi, 0,1,2,3,4,5,6,7,8,9,10,11,12,13,14,15);
}

// Async global->LDS copy of 16 bytes per lane (gfx1250 ASYNCcnt path).
// vdst = LDS byte address, vaddr = 64-bit global address.
static __device__ __forceinline__ void async_copy16(uint32_t lds_off, const void* gptr) {
  asm volatile("global_load_async_to_lds_b128 %0, %1, off"
               :: "v"(lds_off), "v"(gptr) : "memory");
}
static __device__ __forceinline__ void wait_async0() {
  asm volatile("s_wait_asynccnt 0x0" ::: "memory");
}
static __device__ __forceinline__ uint32_t lds_addr(const void* p) {
  // generic LDS address: low 32 bits are the LDS byte address
  return (uint32_t)(uintptr_t)p;
}

// ---------------- fp32 -> bf16 convert (4 elems/thread) ----------------
__global__ void cvt_bf16_kernel(const float* __restrict__ in,
                                __bf16* __restrict__ out, int n) {
  int i = (blockIdx.x * blockDim.x + threadIdx.x) * 4;
  if (i + 3 < n) {
    float4 v = *(const float4*)(in + i);
    v4bf o;
    o[0] = (__bf16)v.x; o[1] = (__bf16)v.y; o[2] = (__bf16)v.z; o[3] = (__bf16)v.w;
    *(v4bf*)(out + i) = o;
  }
}

// ---------------- 64x64 LDS-tiled fp32->bf16 transpose: out[c][r] = in[r][c] --------------
static __device__ __forceinline__ void tile_transpose64(const float* __restrict__ in, int istride,
                                                        __bf16* __restrict__ out, int ostride) {
  __shared__ __bf16 sh[64][72];  // padded to dodge bank conflicts
  const int t = threadIdx.x;
#pragma unroll
  for (int i = 0; i < 4; ++i) {
    int ch = t + i * 256;            // 1024 float4 chunks
    int r = ch >> 4, c4 = (ch & 15) * 4;
    float4 v = *(const float4*)(in + (size_t)r * istride + c4);
    sh[r][c4 + 0] = (__bf16)v.x;
    sh[r][c4 + 1] = (__bf16)v.y;
    sh[r][c4 + 2] = (__bf16)v.z;
    sh[r][c4 + 3] = (__bf16)v.w;
  }
  __syncthreads();
#pragma unroll
  for (int i = 0; i < 2; ++i) {
    int ch = t + i * 256;            // 512 chunks of 8 bf16
    int c = ch >> 3, r8 = (ch & 7) * 8;
    v8bf o;
#pragma unroll
    for (int e = 0; e < 8; ++e) o[e] = sh[r8 + e][c];
    *(v8bf*)(out + (size_t)c * ostride + r8) = o;
  }
}

// per-head transpose of V: YbT[b][h][d][m] = Y[b][m][h*64+d]
__global__ void transpose_head_kernel(const float* __restrict__ Y,
                                      __bf16* __restrict__ YbT) {
  int blk = blockIdx.x;              // 2048 = 128 (b,h) * 16 m-tiles
  int mt = blk & 15;
  int bh = blk >> 4;
  int b = bh >> 4, h = bh & 15;
  const float* in = Y + ((size_t)(b * NSEQ + mt * 64)) * DIM + h * DH;
  __bf16* out = YbT + (size_t)bh * DH * NSEQ + mt * 64;
  tile_transpose64(in, DIM, out, NSEQ);
}

// W1 transpose: W1T[j][k] = W1[k][j]
__global__ void transpose_w1_kernel(const float* __restrict__ W1,
                                    __bf16* __restrict__ W1T) {
  int blk = blockIdx.x;              // 256 = 16 jt * 16 kt
  int kt = blk & 15, jt = blk >> 4;
  const float* in = W1 + (size_t)(kt * 64) * DIM + jt * 64;
  __bf16* out = W1T + (size_t)(jt * 64) * DIM + kt * 64;
  tile_transpose64(in, DIM, out, DIM);
}

// ---------------- fused flash attention + residual: Hpre = X + softmax(QK^T/sqrt(dim)) V --------
// grid: 1024 blocks x 256 threads (8 waves). Wave owns one (b,h,16-query tile);
// all 8 waves share the (b,h) K/V tiles, staged once per WG via async LDS loads.
// Computes S^T = K . Q^T so both score operands load contiguously; each lane
// owns one query column -> online softmax stats are lane-local scalars.
__global__ __launch_bounds__(256, 2)
void flash_kernel(const __bf16* __restrict__ Xb,
                  const __bf16* __restrict__ Yb,
                  const __bf16* __restrict__ YbT,
                  const float*  __restrict__ X,
                  float* __restrict__ Hpre) {
  __shared__ __attribute__((aligned(32))) __bf16 shK[32 * 64];  // K tile, row stride 64
  __shared__ __attribute__((aligned(32))) __bf16 shV[64 * 32];  // V^T tile, row stride 32

  const int t    = threadIdx.x;
  const int wave = t >> 5;
  const int lane = t & 31;
  const int ln   = lane & 15;
  const int half = lane >> 4;

  const int wg  = blockIdx.x;          // 0..1023
  const int qb2 = wg & 7;
  const int h   = (wg >> 3) & 15;
  const int b   = wg >> 7;
  const int q0  = (qb2 * 8 + wave) * 16;

  // per-thread async staging assignment (16B chunks; 4KB K + 4KB V per step)
  const int krow = t >> 3, kch = (t & 7) * 8;
  const int vrow = t >> 2, vch = (t & 3) * 8;
  const __bf16* Kg = Yb  + ((size_t)(b * NSEQ + krow)) * DIM + h * DH + kch;
  const __bf16* Vg = YbT + ((size_t)(b * NH + h) * DH + vrow) * NSEQ + vch;
  const uint32_t kofs = lds_addr(&shK[krow * 64 + kch]);
  const uint32_t vofs = lds_addr(&shV[vrow * 32 + vch]);

  // Q^T B-fragments (fixed for whole kv loop)
  const __bf16* Xrow = Xb + ((size_t)(b * NSEQ + q0)) * DIM + h * DH;
  v16bf Bq0 = load_frag_b(Xrow + 0,  DIM);
  v16bf Bq1 = load_frag_b(Xrow + 32, DIM);

  v8f acc0 = {}, acc1 = {}, acc2 = {}, acc3 = {};
  float mrun = -__builtin_inff();
  float lrun = 0.f;
  const float scale = 0.03125f;  // 1/sqrt(1024)

  for (int kv0 = 0; kv0 < NSEQ; kv0 += 32) {
    __syncthreads();                       // protect LDS from previous step's readers
    async_copy16(kofs, Kg + (size_t)kv0 * DIM);
    async_copy16(vofs, Vg + kv0);
    wait_async0();
    __syncthreads();                       // publish staged K/V to all waves

    v16bf A0c0 = load_frag_a(shK + 0,            64);
    v16bf A0c1 = load_frag_a(shK + 32,           64);
    v16bf A1c0 = load_frag_a(shK + 16 * 64,      64);
    v16bf A1c1 = load_frag_a(shK + 16 * 64 + 32, 64);

    v8f s0 = {}, s1 = {};
    s0 = wmma_bf16(A0c0, Bq0, s0);
    s0 = wmma_bf16(A0c1, Bq1, s0);
    s1 = wmma_bf16(A1c0, Bq0, s1);
    s1 = wmma_bf16(A1c1, Bq1, s1);

    float mloc = -__builtin_inff();
#pragma unroll
    for (int e = 0; e < 8; ++e) {
      s0[e] *= scale; s1[e] *= scale;
      mloc = fmaxf(mloc, fmaxf(s0[e], s1[e]));
    }
    mloc = fmaxf(mloc, __shfl_xor(mloc, 16));
    float mnew = fmaxf(mrun, mloc);
    float corr = __expf(mrun - mnew);

    float p0[8], p1[8];
    float lloc = 0.f;
#pragma unroll
    for (int e = 0; e < 8; ++e) {
      p0[e] = __expf(s0[e] - mnew);
      p1[e] = __expf(s1[e] - mnew);
      lloc += p0[e] + p1[e];
    }
    lloc += __shfl_xor(lloc, 16);
    lrun = lrun * corr + lloc;
    mrun = mnew;

    // rescale running output: factor per query row, broadcast from stat lane
#pragma unroll
    for (int m = 0; m < 8; ++m) {
      float f = __shfl(corr, m + 8 * half);
      acc0[m] *= f; acc1[m] *= f; acc2[m] *= f; acc3[m] *= f;
    }

    // P (16x32) A-fragment: lane-local repack of exp'd scores
    v16bf Pa;
#pragma unroll
    for (int e = 0; e < 8; ++e) {
      Pa[e]     = (__bf16)p0[e];
      Pa[e + 8] = (__bf16)p1[e];
    }

    // O += P . V  (V^T rows = head dims, staged in LDS)
    acc0 = wmma_bf16(Pa, load_frag_b(shV +  0 * 32, 32), acc0);
    acc1 = wmma_bf16(Pa, load_frag_b(shV + 16 * 32, 32), acc1);
    acc2 = wmma_bf16(Pa, load_frag_b(shV + 32 * 32, 32), acc2);
    acc3 = wmma_bf16(Pa, load_frag_b(shV + 48 * 32, 32), acc3);
  }

  // finalize: divide by softmax denominator, add residual X, store Hpre
  float invl[8];
#pragma unroll
  for (int m = 0; m < 8; ++m) {
    float lv = __shfl(lrun, m + 8 * half);
    invl[m] = 1.f / lv;
  }
#pragma unroll
  for (int j = 0; j < 4; ++j) {
    int col = h * DH + j * 16 + ln;
#pragma unroll
    for (int m = 0; m < 8; ++m) {
      int row = q0 + m + 8 * half;
      size_t idx = (size_t)(b * NSEQ + row) * DIM + col;
      float av = (j == 0 ? acc0[m] : j == 1 ? acc1[m] : j == 2 ? acc2[m] : acc3[m]);
      Hpre[idx] = X[idx] + av * invl[m];
    }
  }
}

// ---------------- row LayerNorm (dim=1024, 256 threads, 4 elems/thread) ----------------
__global__ void ln_kernel(const float* __restrict__ in,
                          const float* __restrict__ gamma,
                          const float* __restrict__ beta,
                          float* __restrict__ out_f32,
                          __bf16* __restrict__ out_bf16) {
  const int row = blockIdx.x;
  const float* r = in + (size_t)row * DIM;
  float v[4];
  float s = 0.f, ss = 0.f;
#pragma unroll
  for (int i = 0; i < 4; ++i) {
    v[i] = r[threadIdx.x + 256 * i];
    s += v[i];
    ss += v[i] * v[i];
  }
#pragma unroll
  for (int off = 16; off >= 1; off >>= 1) {
    s  += __shfl_xor(s,  off);
    ss += __shfl_xor(ss, off);
  }
  __shared__ float as_[8], ass_[8];
  int wave = threadIdx.x >> 5;
  if ((threadIdx.x & 31) == 0) { as_[wave] = s; ass_[wave] = ss; }
  __syncthreads();
  float S = 0.f, SS = 0.f;
#pragma unroll
  for (int w = 0; w < 8; ++w) { S += as_[w]; SS += ass_[w]; }
  float mu  = S * (1.f / DIM);
  float var = SS * (1.f / DIM) - mu * mu;
  float rs  = rsqrtf(var + 1e-5f);
#pragma unroll
  for (int i = 0; i < 4; ++i) {
    int c = threadIdx.x + 256 * i;
    float o = (v[i] - mu) * rs * gamma[c] + beta[c];
    out_f32[(size_t)row * DIM + c] = o;
    if (out_bf16) out_bf16[(size_t)row * DIM + c] = (__bf16)o;
  }
}

// ---------------- FFN GEMM: Opre = H + relu(Hb . W1 + b1) ----------------
// grid: 512 blocks x 256 threads. Block owns 16 rows; wave w owns cols [128w,128w+128).
__global__ __launch_bounds__(256, 2)
void ffn_kernel(const __bf16* __restrict__ Hb,
                const __bf16* __restrict__ W1T,
                const float*  __restrict__ b1,
                const float*  __restrict__ H,
                float* __restrict__ Opre) {
  const int wave = threadIdx.x >> 5;
  const int lane = threadIdx.x & 31;
  const int ln   = lane & 15;
  const int half = lane >> 4;
  const int row0 = blockIdx.x * 16;

  v8f acc[8] = {};
  for (int k0 = 0; k0 < DIM; k0 += 32) {
    v16bf A = load_frag_a(Hb + (size_t)row0 * DIM + k0, DIM);
#pragma unroll
    for (int j = 0; j < 8; ++j) {
      v16bf Bf = load_frag_b(W1T + (size_t)(wave * 128 + j * 16) * DIM + k0, DIM);
      acc[j] = wmma_bf16(A, Bf, acc[j]);
    }
  }
#pragma unroll
  for (int j = 0; j < 8; ++j) {
    int col = wave * 128 + j * 16 + ln;
    float bias = b1[col];
#pragma unroll
    for (int m = 0; m < 8; ++m) {
      int row = row0 + m + 8 * half;
      size_t idx = (size_t)row * DIM + col;
      float x = acc[j][m] + bias;
      Opre[idx] = fmaxf(x, 0.f) + H[idx];
    }
  }
}

extern "C" void kernel_launch(void* const* d_in, const int* in_sizes, int n_in,
                              void* d_out, int out_size, void* d_ws, size_t ws_size,
                              hipStream_t stream) {
  const float* X       = (const float*)d_in[0];
  const float* Y       = (const float*)d_in[1];
  const float* W1      = (const float*)d_in[2];
  const float* b1      = (const float*)d_in[3];
  const float* gamma_h = (const float*)d_in[4];
  const float* beta_h  = (const float*)d_in[5];
  const float* gamma_o = (const float*)d_in[6];
  const float* beta_o  = (const float*)d_in[7];
  (void)in_sizes; (void)n_in; (void)out_size; (void)ws_size;

  const size_t MB = (size_t)1 << 20;
  char* ws = (char*)d_ws;
  __bf16* Xb   = (__bf16*)(ws + 0 * MB);    // 16 MB
  __bf16* Yb   = (__bf16*)(ws + 16 * MB);   // 16 MB
  __bf16* YbT  = (__bf16*)(ws + 32 * MB);   // 16 MB
  __bf16* W1T  = (__bf16*)(ws + 48 * MB);   //  2 MB
  __bf16* Hb   = (__bf16*)(ws + 50 * MB);   // 16 MB
  float*  Hpre = (float*)(ws + 66 * MB);    // 32 MB (reused as Opre)
  float*  H    = (float*)(ws + 98 * MB);    // 32 MB
  float*  Opre = Hpre;
  float*  out  = (float*)d_out;

  const int NTOK = NB * NSEQ * DIM;  // 8388608

  cvt_bf16_kernel<<<NTOK / 1024, 256, 0, stream>>>(X, Xb, NTOK);
  cvt_bf16_kernel<<<NTOK / 1024, 256, 0, stream>>>(Y, Yb, NTOK);
  transpose_head_kernel<<<NB * NH * 16, 256, 0, stream>>>(Y, YbT);      // 2048 blocks
  transpose_w1_kernel<<<256, 256, 0, stream>>>(W1, W1T);

  flash_kernel<<<1024, 256, 0, stream>>>(Xb, Yb, YbT, X, Hpre);

  ln_kernel<<<NB * NSEQ, 256, 0, stream>>>(Hpre, gamma_h, beta_h, H, Hb);

  ffn_kernel<<<(NB * NSEQ) / 16, 256, 0, stream>>>(Hb, W1T, b1, H, Opre);

  ln_kernel<<<NB * NSEQ, 256, 0, stream>>>(Opre, gamma_o, beta_o, out, nullptr);
}
